// PIN_30528627540805
// MI455X (gfx1250) — matrix operand order repaired and checked by reading, not compiled, not run on hardware
//
#include <hip/hip_runtime.h>

// ---------------------------------------------------------------------------
// TRADE-style decoder step for MI455X (gfx1250, wave32).
// All heavy GEMMs run through v_wmma_f32_16x16x32_bf16 with f32 accumulation.
// Staging buffers are bf16 with the K dimension zero-padded to a multiple of
// 32 and BOTH the A-row and Bt-row counts padded to the full WMMA tile grid,
// so the GEMM K-loop is completely branch-free: 10x global_load_b128 + 4x
// v_wmma per 32-K chunk, no EXEC manipulation, no spills.
// ---------------------------------------------------------------------------

typedef __attribute__((ext_vector_type(16))) __bf16 v16bf;
typedef __attribute__((ext_vector_type(8)))  float  v8f;

union FragBF {                 // natural alignment 32 (from v16bf)
    v16bf v;
    unsigned u[8];
    uint4 q[2];
};

__device__ __forceinline__ unsigned short f32_to_bf16_bits(float f) {
    unsigned u = __float_as_uint(f);
    u += 0x7FFFu + ((u >> 16) & 1u);   // round-to-nearest-even
    return (unsigned short)(u >> 16);
}

// ---------------------------------------------------------------------------
// Batched GEMM  C[M,N] = A[M,K] * Bt[N,K]^T   (bf16 in, f32 out)
//  - lda/ldb element strides (mult of 32); Kp = K padded to 32, zero-filled.
//  - A allocation covers ceil16(M) rows; Bt allocation covers the full
//    NB*16-column tile grid, so NO load guards are needed: out-of-range rows
//    only feed accumulator elements the store discards.
//  - One wave computes a 16 x (16*NB) strip; A fragment reused NB times.
//  - Fragment layouts per CDNA5 ISA (16-bit A 16x32 / B 32x16):
//    both reduce to two contiguous 16B loads per lane.
// ---------------------------------------------------------------------------
template <int NB>
__global__ void gemm_bf16_wmma(const unsigned short* __restrict__ A, long sAb, int lda,
                               const unsigned short* __restrict__ B, long sBb, int ldb,
                               float* __restrict__ C, long sCb, int ldc,
                               int M, int N, int Kp) {
    const int batch = blockIdx.y;
    A += (long)batch * sAb;
    B += (long)batch * sBb;
    C += (long)batch * sCb;

    const int lane = threadIdx.x & 31;
    // readfirstlane -> wave coordinates live in SGPRs; all tile math scalar
    const int wave = __builtin_amdgcn_readfirstlane(threadIdx.x >> 5);
    const int group = blockIdx.x * (blockDim.x >> 5) + wave;

    const int ntg = (N + 16 * NB - 1) / (16 * NB);
    const int mt  = group / ntg;
    const int nt0 = (group % ntg) * NB;
    if (mt * 16 >= M) return;            // scalar branch

    const int row0 = mt * 16;
    const int l15  = lane & 15;          // A row-in-tile / Bt col-in-tile
    const int g    = lane >> 4;          // half-wave selector

    v8f acc[NB];
    v8f zero = {0.f, 0.f, 0.f, 0.f, 0.f, 0.f, 0.f, 0.f};
#pragma unroll
    for (int i = 0; i < NB; ++i) acc[i] = zero;

    const unsigned short* aRow = A + (long)(row0 + l15) * lda + 8 * g;
    const unsigned short* bCol = B + (long)(nt0 * 16 + l15) * ldb + 8 * g;

    for (int k0 = 0; k0 < Kp; k0 += 32) {
        // A fragment: VGPRs 0-3 hold k = 8g..8g+7, VGPRs 4-7 hold k = 16+8g..
        FragBF a;
        a.q[0] = *(const uint4*)(aRow + k0);
        a.q[1] = *(const uint4*)(aRow + k0 + 16);
#pragma unroll
        for (int i = 0; i < NB; ++i) {
            // B fragment: same k order as A (k = 8g..8g+7, then 16+8g..16+8g+7)
            FragBF b;
            const unsigned short* bp = bCol + (long)i * 16 * ldb + k0;
            b.q[0] = *(const uint4*)(bp);
            b.q[1] = *(const uint4*)(bp + 16);
            acc[i] = __builtin_amdgcn_wmma_f32_16x16x32_bf16(
                false, a.v, false, b.v, (short)0, acc[i], false, false);
        }
    }

    // store: VGPR r holds (M = row0 + r + 8g, N = col0 + l15)
#pragma unroll
    for (int i = 0; i < NB; ++i) {
        const int col0 = (nt0 + i) * 16;
        if (col0 >= N) continue;         // scalar guard (store stage only)
        const int n = col0 + l15;
        if (n < N) {
#pragma unroll
            for (int r = 0; r < 8; ++r) {
                const int mm = row0 + r + 8 * g;
                if (mm < M) C[(long)mm * ldc + n] = acc[i][r];
            }
        }
    }
}

// ---------------------------------------------------------------------------
// Staging / pointwise kernels
// ---------------------------------------------------------------------------

// rows x K (f32, dense) -> rows x Kp (bf16, zero pad in [K,Kp))
__global__ void cast_pad_bf16(const float* __restrict__ in,
                              unsigned short* __restrict__ out,
                              long rows, int K, int Kp) {
    long i = blockIdx.x * (long)blockDim.x + threadIdx.x;
    if (i >= rows * Kp) return;
    int k = (int)(i % Kp);
    long r = i / Kp;
    out[i] = (k < K) ? f32_to_bf16_bits(in[r * K + k]) : (unsigned short)0;
}

// (B,L,H) f32 -> (B, rowPad, Hp) bf16 rows 0..L-1 (K pad zeroed)
__global__ void seq_cast_bf16(const float* __restrict__ in,
                              unsigned short* __restrict__ out,
                              int Bsz, int L, int H, int Hp, int rowPad) {
    long i = blockIdx.x * (long)blockDim.x + threadIdx.x;
    if (i >= (long)Bsz * L * Hp) return;
    int k = (int)(i % Hp);
    long rem = i / Hp;
    int l = (int)(rem % L);
    int b = (int)(rem / L);
    out[((long)b * rowPad + l) * Hp + k] =
        (k < H) ? f32_to_bf16_bits(in[((long)b * L + l) * H + k]) : (unsigned short)0;
}

// (B,L,H) f32 -> transposed (B, rowPad, Lp) bf16: out[b][h][l] = in[b][l][h]
__global__ void seq_transpose_bf16(const float* __restrict__ in,
                                   unsigned short* __restrict__ out,
                                   int Bsz, int L, int H, int Lp, int rowPad) {
    long i = blockIdx.x * (long)blockDim.x + threadIdx.x;
    if (i >= (long)Bsz * H * Lp) return;
    int l = (int)(i % Lp);
    long rem = i / Lp;
    int hc = (int)(rem % H);
    int b = (int)(rem / H);
    out[((long)b * rowPad + hc) * Lp + l] =
        (l < L) ? f32_to_bf16_bits(in[((long)b * L + l) * H + hc]) : (unsigned short)0;
}

// out[(s*B+b)*Hp + k] = slot[s*H+k] (bf16, K pad zeroed)
__global__ void bcast_slot_bf16(const float* __restrict__ slot,
                                unsigned short* __restrict__ out,
                                int Bsz, int H, int Hp, long n) {
    long i = blockIdx.x * (long)blockDim.x + threadIdx.x;
    if (i >= n) return;
    int k = (int)(i % Hp);
    int s = (int)((i / Hp) / Bsz);
    out[i] = (k < H) ? f32_to_bf16_bits(slot[(long)s * H + k]) : (unsigned short)0;
}
__global__ void bcast_slot_f32(const float* __restrict__ slot,
                               float* __restrict__ out, int Bsz, int H, long n) {
    long i = blockIdx.x * (long)blockDim.x + threadIdx.x;
    if (i >= n) return;
    int j = (int)(i % H);
    int s = (int)((i / H) / Bsz);
    out[i] = slot[(long)s * H + j];
}

__global__ void add2_kernel(const float* __restrict__ a, const float* __restrict__ b,
                            float* __restrict__ c, long n) {
    long i = blockIdx.x * (long)blockDim.x + threadIdx.x;
    if (i < n) c[i] = a[i] + b[i];
}

// softmax over L per (s,b) row with mask; p is dense f32, p_bf padded to Lp
__global__ void attn_softmax_kernel(const float* __restrict__ scores,
                                    const float* __restrict__ musk,
                                    float* __restrict__ p,
                                    unsigned short* __restrict__ p_bf,
                                    int Bsz, int L, int Lp) {
    const int rid = blockIdx.x;               // s*B + b
    const int b = rid % Bsz;
    const int tid = threadIdx.x;
    const float* srow = scores + (long)rid * L;
    const float* mrow = musk + (long)b * L;
    __shared__ float red[128];

    float mx = -3.4e38f;
    for (int l = tid; l < L; l += blockDim.x) mx = fmaxf(mx, srow[l] + mrow[l]);
    red[tid] = mx; __syncthreads();
    for (int s = blockDim.x / 2; s > 0; s >>= 1) {
        if (tid < s) red[tid] = fmaxf(red[tid], red[tid + s]);
        __syncthreads();
    }
    mx = red[0]; __syncthreads();

    float sum = 0.f;
    for (int l = tid; l < L; l += blockDim.x) sum += __expf(srow[l] + mrow[l] - mx);
    red[tid] = sum; __syncthreads();
    for (int s = blockDim.x / 2; s > 0; s >>= 1) {
        if (tid < s) red[tid] += red[tid + s];
        __syncthreads();
    }
    const float inv = 1.f / red[0];

    for (int l = tid; l < Lp; l += blockDim.x) {
        if (l < L) {
            float v = __expf(srow[l] + mrow[l] - mx) * inv;
            p[(long)rid * L + l] = v;
            p_bf[(long)rid * Lp + l] = f32_to_bf16_bits(v);
        } else {
            p_bf[(long)rid * Lp + l] = 0;   // K pad must be zero
        }
    }
}

// GRU pointwise: h <- (1-z)*n + z*h  (in-place on h)
__global__ void gru_pointwise_kernel(const float* __restrict__ gi,
                                     const float* __restrict__ gh,
                                     const float* __restrict__ b_ih,
                                     const float* __restrict__ b_hh,
                                     float* __restrict__ h, long rows, int H) {
    long idx = blockIdx.x * (long)blockDim.x + threadIdx.x;
    if (idx >= rows * H) return;
    const long rid = idx / H;
    const int  j   = (int)(idx % H);
    const float* gir = gi + rid * 3 * H;
    const float* ghr = gh + rid * 3 * H;
    float ir  = gir[j]         + b_ih[j];
    float iz  = gir[H + j]     + b_ih[H + j];
    float in_ = gir[2 * H + j] + b_ih[2 * H + j];
    float hr  = ghr[j]         + b_hh[j];
    float hz  = ghr[H + j]     + b_hh[H + j];
    float hn  = ghr[2 * H + j] + b_hh[2 * H + j];
    float r = 1.f / (1.f + __expf(-(ir + hr)));
    float z = 1.f / (1.f + __expf(-(iz + hz)));
    float n = tanhf(in_ + r * hn);
    h[idx] = (1.f - z) * n + z * h[idx];
}

// gate = concat(ctx_s,ctx_u) @ Wg^T + b   (only needed for t==0)
__global__ void gate_kernel(const float* __restrict__ ctx_s, const float* __restrict__ ctx_u,
                            const float* __restrict__ Wg_w, const float* __restrict__ Wg_b,
                            float* __restrict__ out_gate, int rows, int H, int G) {
    int idx = blockIdx.x * blockDim.x + threadIdx.x;
    if (idx >= rows * G) return;
    const int gi  = idx % G;
    const int rid = idx / G;
    const float* w = Wg_w + (long)gi * 2 * H;
    float acc = Wg_b[gi];
    for (int j = 0; j < H; ++j)
        acc += ctx_s[(long)rid * H + j] * w[j] + ctx_u[(long)rid * H + j] * w[H + j];
    out_gate[idx] = acc;
}

// per (s,b): sw = sigmoid([h,ctx_s,ctx_u,x]·Wr + b);
// alpha/beta = softmax([h,ctx_s,x]·Wc, [h,ctx_u,x]·Wc);
// store {sw, (1-sw)*alpha, (1-sw)*beta}
__global__ void coef_kernel(const float* __restrict__ h, const float* __restrict__ ctx_s,
                            const float* __restrict__ ctx_u, const float* __restrict__ x,
                            const float* __restrict__ Wr_w, const float* __restrict__ Wr_b,
                            const float* __restrict__ Wc_w, const float* __restrict__ Wc_b,
                            float* __restrict__ coef, int H) {
    const int rid = blockIdx.x;
    const int tid = threadIdx.x;
    const long base = (long)rid * H;
    float sr = 0.f, scs = 0.f, scu = 0.f;
    for (int j = tid; j < H; j += blockDim.x) {
        float hv = h[base + j], cs = ctx_s[base + j], cu = ctx_u[base + j], xv = x[base + j];
        sr  += hv * Wr_w[j] + cs * Wr_w[H + j] + cu * Wr_w[2 * H + j] + xv * Wr_w[3 * H + j];
        scs += hv * Wc_w[j] + cs * Wc_w[H + j] + xv * Wc_w[2 * H + j];
        scu += hv * Wc_w[j] + cu * Wc_w[H + j] + xv * Wc_w[2 * H + j];
    }
    __shared__ float r0[128], r1[128], r2[128];
    r0[tid] = sr; r1[tid] = scs; r2[tid] = scu;
    __syncthreads();
    for (int s = blockDim.x / 2; s > 0; s >>= 1) {
        if (tid < s) { r0[tid] += r0[tid + s]; r1[tid] += r1[tid + s]; r2[tid] += r2[tid + s]; }
        __syncthreads();
    }
    if (tid == 0) {
        float sw = 1.f / (1.f + __expf(-(r0[0] + Wr_b[0])));
        float ls = r1[0] + Wc_b[0], lu = r2[0] + Wc_b[0];
        float m = fmaxf(ls, lu);
        float es = __expf(ls - m), eu = __expf(lu - m);
        float inv = 1.f / (es + eu);
        coef[rid * 3 + 0] = sw;
        coef[rid * 3 + 1] = (1.f - sw) * es * inv;
        coef[rid * 3 + 2] = (1.f - sw) * eu * inv;
    }
}

// row max + 1/sum(exp) over V per (s,b) row of logits (logits stay L2-resident)
__global__ void row_softmax_stats(const float* __restrict__ logits,
                                  float* __restrict__ rmax, float* __restrict__ rinv, int V) {
    const int rid = blockIdx.x;
    const int tid = threadIdx.x;
    const float* row = logits + (long)rid * V;
    __shared__ float red[256];
    float mx = -3.4e38f;
    for (int v = tid; v < V; v += blockDim.x) mx = fmaxf(mx, row[v]);
    red[tid] = mx; __syncthreads();
    for (int s = blockDim.x / 2; s > 0; s >>= 1) {
        if (tid < s) red[tid] = fmaxf(red[tid], red[tid + s]);
        __syncthreads();
    }
    mx = red[0]; __syncthreads();
    float sum = 0.f;
    for (int v = tid; v < V; v += blockDim.x) sum += __expf(row[v] - mx);
    red[tid] = sum; __syncthreads();
    for (int s = blockDim.x / 2; s > 0; s >>= 1) {
        if (tid < s) red[tid] += red[tid + s];
        __syncthreads();
    }
    if (tid == 0) { rmax[rid] = mx; rinv[rid] = 1.f / red[0]; }
}

// out[b,s,t,:] = sw * softmax(logits)  (overwrite; scatter adds come after)
__global__ void final_blend_kernel(const float* __restrict__ logits,
                                   const float* __restrict__ rmax,
                                   const float* __restrict__ rinv,
                                   const float* __restrict__ coef,
                                   float* __restrict__ out,
                                   int S, int Bsz, int T, int t, int V) {
    const int rid = blockIdx.y;               // s*B + b
    const int v = blockIdx.x * blockDim.x + threadIdx.x;
    if (v >= V) return;
    const int s = rid / Bsz, b = rid % Bsz;
    const float sw = coef[rid * 3 + 0];
    const float p = __expf(logits[(long)rid * V + v] - rmax[rid]) * rinv[rid];
    out[(((long)b * S + s) * T + t) * V + v] = sw * p;
}

// copy distribution: atomically add coef * p[s,b,l] into out[..., story[b,l]]
__global__ void copy_scatter_kernel(const float* __restrict__ p,
                                    const int* __restrict__ story,
                                    const float* __restrict__ coef, int which,
                                    float* __restrict__ out,
                                    int S, int Bsz, int T, int t, int L, int V) {
    long idx = blockIdx.x * (long)blockDim.x + threadIdx.x;
    if (idx >= (long)S * Bsz * L) return;
    const int l = (int)(idx % L);
    const int rid = (int)(idx / L);
    const int b = rid % Bsz, s = rid / Bsz;
    const float c = coef[rid * 3 + which];
    const int tok = story[(long)b * L + l];
    atomicAdd(&out[(((long)b * S + s) * T + t) * V + tok], c * p[idx]);
}

// x[s,b,:] = emb[targets[b,s,t], :]
__global__ void gather_x_kernel(const float* __restrict__ emb, const int* __restrict__ targets,
                                float* __restrict__ x, int S, int Bsz, int T, int t, int H) {
    long idx = blockIdx.x * (long)blockDim.x + threadIdx.x;
    if (idx >= (long)S * Bsz * H) return;
    const int j = (int)(idx % H);
    const int rid = (int)(idx / H);
    const int b = rid % Bsz, s = rid / Bsz;
    const int tok = targets[((long)b * S + s) * T + t];
    x[idx] = emb[(long)tok * H + j];
}

// ---------------------------------------------------------------------------
// Host side
// ---------------------------------------------------------------------------
static void launch_gemm(const unsigned short* A, long sAb, int lda,
                        const unsigned short* B, long sBb, int ldb,
                        float* C, long sCb, int ldc,
                        int M, int N, int Kp, int batch, hipStream_t st) {
    constexpr int NB = 4;
    const int ntg = (N + 16 * NB - 1) / (16 * NB);
    const int mtiles = (M + 15) / 16;
    const int groups = mtiles * ntg;
    dim3 grid((groups + 7) / 8, batch);
    gemm_bf16_wmma<NB><<<grid, 256, 0, st>>>(A, sAb, lda, B, sBb, ldb, C, sCb, ldc, M, N, Kp);
}

static inline dim3 grid1d(long n, int bs) { return dim3((unsigned)((n + bs - 1) / bs)); }

extern "C" void kernel_launch(void* const* d_in, const int* in_sizes, int n_in,
                              void* d_out, int out_size, void* d_ws, size_t ws_size,
                              hipStream_t stream) {
    constexpr int B = 16, L = 200, H = 400, V = 20000, S = 30, T = 10, G = 3;
    constexpr long SB = (long)S * B;
    // padded geometry
    constexpr int Hp = 416;      // K pad of H (mult of 32)
    constexpr int Lp = 224;      // K pad of L (mult of 32)
    constexpr int SBp = 512;     // A-row pad for M=480 / M=30-per-batch strips
    constexpr int LrowP = 256;   // Bt-row pad for N=200 (NB=4 -> 64-col groups)
    constexpr int HrowP = 448;   // Bt-row pad for N=400
    constexpr int WrowP = 1216;  // Bt-row pad for N=1200
    constexpr int VrowP = 20032; // Bt-row pad for N=20000

    const float* sys_H    = (const float*)d_in[0];
    const float* user_H   = (const float*)d_in[1];
    const float* sys_musk = (const float*)d_in[2];
    const float* user_musk= (const float*)d_in[3];
    const float* emb      = (const float*)d_in[4];
    const float* W_ih     = (const float*)d_in[5];
    const float* W_hh     = (const float*)d_in[6];
    const float* b_ih     = (const float*)d_in[7];
    const float* b_hh     = (const float*)d_in[8];
    const float* Wg_w     = (const float*)d_in[9];
    const float* Wg_b     = (const float*)d_in[10];
    const float* Wr_w     = (const float*)d_in[11];
    const float* Wr_b     = (const float*)d_in[12];
    const float* Wc_w     = (const float*)d_in[13];
    const float* Wc_b     = (const float*)d_in[14];
    const float* slot_att = (const float*)d_in[15];
    const float* slot_emb = (const float*)d_in[16];
    const int*   story_sys  = (const int*)d_in[17];
    const int*   story_user = (const int*)d_in[18];
    const int*   targets    = (const int*)d_in[19];

    float* out_points = (float*)d_out;                          // (B,S,T,V)
    float* out_gates  = (float*)d_out + (long)B * S * T * V;    // (S,B,G)

    // ---- carve workspace (~80 MB)
    char* wp = (char*)d_ws;
    auto carve = [&](size_t bytes) -> void* {
        void* r = (void*)wp;
        wp += (bytes + 255) & ~(size_t)255;
        return r;
    };
    unsigned short* embB  = (unsigned short*)carve((size_t)VrowP * Hp * 2);
    unsigned short* sysB  = (unsigned short*)carve((size_t)B * LrowP * Hp * 2);
    unsigned short* userB = (unsigned short*)carve((size_t)B * LrowP * Hp * 2);
    unsigned short* sysT  = (unsigned short*)carve((size_t)B * HrowP * Lp * 2);
    unsigned short* userT = (unsigned short*)carve((size_t)B * HrowP * Lp * 2);
    unsigned short* wihB  = (unsigned short*)carve((size_t)WrowP * Hp * 2);
    unsigned short* whhB  = (unsigned short*)carve((size_t)WrowP * Hp * 2);
    float*          h     = (float*)carve(SB * H * 4);
    unsigned short* hB    = (unsigned short*)carve((size_t)SBp * Hp * 2);
    float*          x     = (float*)carve(SB * H * 4);
    unsigned short* xB    = (unsigned short*)carve((size_t)SBp * Hp * 2);
    float*          gi    = (float*)carve(SB * 3 * H * 4);
    float*          gh    = (float*)carve(SB * 3 * H * 4);
    float*          scores= (float*)carve(SB * L * 4);
    float*          p_s   = (float*)carve(SB * L * 4);
    float*          p_u   = (float*)carve(SB * L * 4);
    unsigned short* psB   = (unsigned short*)carve((size_t)SBp * Lp * 2);
    unsigned short* puB   = (unsigned short*)carve((size_t)SBp * Lp * 2);
    float*          ctx_s = (float*)carve(SB * H * 4);
    float*          ctx_u = (float*)carve(SB * H * 4);
    float*          coef  = (float*)carve(SB * 3 * 4);
    float*          logits= (float*)carve(SB * V * 4);
    float*          rmax  = (float*)carve(SB * 4);
    float*          rinv  = (float*)carve(SB * 4);
    (void)ws_size; (void)in_sizes; (void)n_in; (void)out_size;

    // ---- one-time bf16 staging (K-padded / transposed)
    cast_pad_bf16<<<grid1d((long)V * Hp, 256), 256, 0, stream>>>(emb, embB, V, H, Hp);
    cast_pad_bf16<<<grid1d((long)3 * H * Hp, 256), 256, 0, stream>>>(W_ih, wihB, 3 * H, H, Hp);
    cast_pad_bf16<<<grid1d((long)3 * H * Hp, 256), 256, 0, stream>>>(W_hh, whhB, 3 * H, H, Hp);
    seq_cast_bf16<<<grid1d((long)B * L * Hp, 256), 256, 0, stream>>>(sys_H,  sysB,  B, L, H, Hp, LrowP);
    seq_cast_bf16<<<grid1d((long)B * L * Hp, 256), 256, 0, stream>>>(user_H, userB, B, L, H, Hp, LrowP);
    seq_transpose_bf16<<<grid1d((long)B * H * Lp, 256), 256, 0, stream>>>(sys_H,  sysT,  B, L, H, Lp, HrowP);
    seq_transpose_bf16<<<grid1d((long)B * H * Lp, 256), 256, 0, stream>>>(user_H, userT, B, L, H, Lp, HrowP);

    // ---- initial attention with cond0 = slot_att (broadcast over batch)
    bcast_slot_bf16<<<grid1d(SB * Hp, 256), 256, 0, stream>>>(slot_att, hB, B, H, Hp, SB * Hp);
    bcast_slot_f32 <<<grid1d(SB * H, 256), 256, 0, stream>>>(slot_emb, x, B, H, SB * H);

    auto attend = [&](const unsigned short* seqB, const unsigned short* seqT,
                      const float* musk, float* p, unsigned short* pB, float* ctx) {
        // scores[s,b,l] = sum_k hB[s,b,k] * seqB[b,l,k]
        launch_gemm(hB, (long)Hp, B * Hp,
                    seqB, (long)LrowP * Hp, Hp,
                    scores, (long)L, B * L,
                    S, L, Hp, B, stream);
        attn_softmax_kernel<<<(int)SB, 128, 0, stream>>>(scores, musk, p, pB, B, L, Lp);
        // ctx[s,b,k] = sum_l p[s,b,l] * seqT[b,k,l]
        launch_gemm(pB, (long)Lp, B * Lp,
                    seqT, (long)HrowP * Lp, Lp,
                    ctx, (long)H, B * H,
                    S, H, Lp, B, stream);
    };

    attend(sysB,  sysT,  sys_musk,  p_s, psB, ctx_s);
    attend(userB, userT, user_musk, p_u, puB, ctx_u);
    add2_kernel<<<grid1d(SB * H, 256), 256, 0, stream>>>(ctx_s, ctx_u, h, SB * H);

    // ---- T-step scan (sequential dependence through h only)
    for (int t = 0; t < T; ++t) {
        // GRU cell: gi = x@W_ih^T, gh = h@W_hh^T, then pointwise
        cast_pad_bf16<<<grid1d(SB * Hp, 256), 256, 0, stream>>>(x, xB, SB, H, Hp);
        cast_pad_bf16<<<grid1d(SB * Hp, 256), 256, 0, stream>>>(h, hB, SB, H, Hp);
        launch_gemm(xB, 0, Hp, wihB, 0, Hp, gi, 0, 3 * H, (int)SB, 3 * H, Hp, 1, stream);
        launch_gemm(hB, 0, Hp, whhB, 0, Hp, gh, 0, 3 * H, (int)SB, 3 * H, Hp, 1, stream);
        gru_pointwise_kernel<<<grid1d(SB * H, 256), 256, 0, stream>>>(gi, gh, b_ih, b_hh, h, SB, H);

        // attention with new h
        cast_pad_bf16<<<grid1d(SB * Hp, 256), 256, 0, stream>>>(h, hB, SB, H, Hp);
        attend(sysB,  sysT,  sys_musk,  p_s, psB, ctx_s);
        attend(userB, userT, user_musk, p_u, puB, ctx_u);

        // gate output only consumed at t==0 (reference returns gates[0])
        if (t == 0)
            gate_kernel<<<grid1d(SB * G, 128), 128, 0, stream>>>(ctx_s, ctx_u, Wg_w, Wg_b,
                                                                 out_gates, (int)SB, H, G);

        // sw / alpha / beta coefficients
        coef_kernel<<<(int)SB, 128, 0, stream>>>(h, ctx_s, ctx_u, x,
                                                 Wr_w, Wr_b, Wc_w, Wc_b, coef, H);

        // vocab projection (dominant GEMM; logits stay in the 192MB L2)
        launch_gemm(hB, 0, Hp, embB, 0, Hp, logits, 0, V, (int)SB, V, Hp, 1, stream);
        row_softmax_stats<<<(int)SB, 256, 0, stream>>>(logits, rmax, rinv, V);

        // fused softmax-scale + store of generation part, then copy scatters
        {
            dim3 grid((V + 255) / 256, (unsigned)SB);
            final_blend_kernel<<<grid, 256, 0, stream>>>(logits, rmax, rinv, coef,
                                                         out_points, S, B, T, t, V);
        }
        copy_scatter_kernel<<<grid1d(SB * L, 256), 256, 0, stream>>>(
            p_s, story_sys, coef, 1, out_points, S, B, T, t, L, V);
        copy_scatter_kernel<<<grid1d(SB * L, 256), 256, 0, stream>>>(
            p_u, story_user, coef, 2, out_points, S, B, T, t, L, V);

        // next x = emb[targets[:, :, t]]
        if (t + 1 < T)
            gather_x_kernel<<<grid1d(SB * H, 256), 256, 0, stream>>>(emb, targets, x, S, B, T, t, H);
    }
}
